// KerasSaSentTensorflow_62663572848647
// MI455X (gfx1250) — matrix operand order repaired and checked by reading, not compiled, not run on hardware
//
#include <hip/hip_runtime.h>
#include <hip/hip_bf16.h>

typedef __attribute__((ext_vector_type(16))) _Float16 v16h;
typedef __attribute__((ext_vector_type(8)))  float    v8f;

// ---------------- problem constants ----------------
constexpr int B_  = 64;
constexpr int T_  = 200;
constexpr int WD_ = 300;
constexpr int MD_ = 50;
constexpr int H_  = 256;
constexpr int NC_ = 3;
constexpr int D_  = WD_ + MD_;   // 350
constexpr int G_  = 4 * H_;      // 1024 (gates)
constexpr int M_  = B_ * T_;     // 12800
constexpr int KX_ = 512;         // x-buffer row stride (covers both layers)
constexpr int K0_ = 352;         // layer0 K padded to mult of 32
constexpr int NR_ = 384;         // residual-out cols padded to mult of 64

// ---------------- workspace layout (all sizes multiples of 256B) ----------------
constexpr size_t SZ_EMB = (size_t)M_ * D_  * 4;   // fp32 input embedding
constexpr size_t SZ_X   = (size_t)M_ * KX_ * 2;   // f16 layer input (per buffer)
constexpr size_t SZ_XP  = (size_t)M_ * G_  * 2;   // f16 x-projection (per dir)
constexpr size_t SZ_HS  = (size_t)M_ * H_  * 4;   // f32 h sequence (per dir)
constexpr size_t SZ_WXT = (size_t)G_ * KX_ * 2;   // f16 WxT (per dir)
constexpr size_t SZ_WHT = (size_t)G_ * H_  * 2;   // f16 WhT (per dir)
constexpr size_t SZ_RWT = (size_t)NR_ * KX_ * 2;  // f16 resW transposed
constexpr size_t SZ_LO  = (size_t)M_ * D_ * 4;    // f32 lstm_output

constexpr size_t O_EMB  = 0;
constexpr size_t O_X    = O_EMB + SZ_EMB;            // 2 buffers (fwd, rev)
constexpr size_t O_XP   = O_X   + 2 * SZ_X;          // 2 dirs
constexpr size_t O_HS   = O_XP  + 2 * SZ_XP;         // 2 dirs
constexpr size_t O_WXT  = O_HS  + 2 * SZ_HS;         // 2 dirs
constexpr size_t O_WHT  = O_WXT + 2 * SZ_WXT;        // 2 dirs
constexpr size_t O_RWT  = O_WHT + 2 * SZ_WHT;
constexpr size_t O_LO   = O_RWT + SZ_RWT;
constexpr size_t O_CIN  = O_LO  + SZ_LO;             // f32 [M][2]
constexpr size_t O_COUT = O_CIN + (size_t)M_ * 2 * 4;// f32 [M]
constexpr size_t O_SENT = O_COUT+ (size_t)M_ * 4;    // f32 [B][D]
constexpr size_t O_CST  = O_SENT+ (size_t)B_ * D_ * 4;     // f32 c state, 2 dirs
constexpr size_t O_HST  = O_CST + 2 * (size_t)B_ * H_ * 4; // f32 h state, 2 dirs
constexpr size_t O_H16  = O_HST + 2 * (size_t)B_ * H_ * 4; // f16 h state, 2 dirs x 2 parity
constexpr size_t WS_TOTAL = O_H16 + 4 * (size_t)B_ * H_ * 2;

// ---------------- device helpers ----------------
__device__ __forceinline__ float sigm(float x) { return 1.0f / (1.0f + expf(-x)); }

// A-matrix 16x32 f16 tile load (row-major src): lane&15 = row,
// lane>=16 selects K chunks {8..15,24..31} vs {0..7,16..23}.
__device__ __forceinline__ v16h ldA(const _Float16* base, int ld, int r0, int k0, int lane) {
  const _Float16* p = base + (size_t)(r0 + (lane & 15)) * ld + k0 + ((lane >> 4) << 3);
  v16h v;
#pragma unroll
  for (int j = 0; j < 8; ++j) { v[j] = p[j]; v[j + 8] = p[j + 16]; }
  return v;
}

// B-matrix 32x16 f16 tile load from transposed weights Wt[n][k] (row-major,
// ld = K stride): lane&15 = column n, lanes 0-15 hold K chunks {0..7,16..23},
// lanes 16-31 hold K chunks {8..15,24..31} (same order as ldA).
__device__ __forceinline__ v16h ldB(const _Float16* base, int ld, int n0, int k0, int lane) {
  const _Float16* p = base + (size_t)(n0 + (lane & 15)) * ld + k0 + ((lane >> 4) << 3);
  v16h v;
#pragma unroll
  for (int j = 0; j < 8; ++j) { v[j] = p[j]; v[j + 8] = p[j + 16]; }
  return v;
}

// ---------------- kernels ----------------

// Gather embeddings -> fp32 emb, f16 forward input, f16 reversed input.
__global__ void k_embed(const int* __restrict__ tokens, const int* __restrict__ masks,
                        const int* __restrict__ length,
                        const float* __restrict__ emb, const float* __restrict__ memb,
                        float* __restrict__ emb32, _Float16* __restrict__ xF,
                        _Float16* __restrict__ xR) {
  int idx = blockIdx.x * blockDim.x + threadIdx.x;
  if (idx >= B_ * T_ * K0_) return;
  int d  = idx % K0_;
  int bt = idx / K0_;
  int t  = bt % T_;
  int b  = bt / T_;
  int tok = tokens[b * T_ + t];
  int mk  = masks[b * T_ + t];
  float v;
  if (d < WD_)      v = emb[(size_t)tok * WD_ + d];
  else if (d < D_)  v = memb[mk * MD_ + (d - WD_)];
  else              v = 0.0f;
  if (d < D_) emb32[((size_t)b * T_ + t) * D_ + d] = v;
  int L  = length[b];
  int tr = (t < L) ? (L - 1 - t) : t;   // rev_idx (involution)
  _Float16 hv = (_Float16)v;
  xF[((size_t)b * T_ + t)  * KX_ + d] = hv;
  xR[((size_t)b * T_ + tr) * KX_ + d] = hv;
}

// Split + transpose + f16-convert LSTM weights: W (Din+H, 4H) fp32 ->
// WxT[dir][n][k] (k<KX, zero padded) and WhT[dir][n][k] (k<H).
__global__ void k_wprep(const float* __restrict__ Wf, const float* __restrict__ Wb,
                        int Din, _Float16* __restrict__ wxT, _Float16* __restrict__ whT) {
  int idx = blockIdx.x * blockDim.x + threadIdx.x;
  if (idx >= 2 * G_ * KX_) return;
  int k   = idx % KX_;
  int n   = (idx / KX_) % G_;
  int dir = idx / (KX_ * G_);
  const float* W = dir ? Wb : Wf;
  float vx = (k < Din) ? W[(size_t)k * G_ + n] : 0.0f;
  wxT[(size_t)dir * G_ * KX_ + (size_t)n * KX_ + k] = (_Float16)vx;
  if (k < H_) {
    float vh = W[(size_t)(Din + k) * G_ + n];
    whT[(size_t)dir * G_ * H_ + (size_t)n * H_ + k] = (_Float16)vh;
  }
}

// Big WMMA GEMM, 16x64 output strip per wave (A-tile reused across 4 B-tiles):
// xproj[dir][m][n] = x[dir][m][:] @ WxT[dir][n][:] + bias[n]  (f16 out)
__global__ __launch_bounds__(128) void k_xproj(const _Float16* __restrict__ x,
                                               const _Float16* __restrict__ wxT,
                                               const float* __restrict__ biasF,
                                               const float* __restrict__ biasB,
                                               _Float16* __restrict__ xproj, int kAct) {
  int dir  = blockIdx.z;
  int wave = threadIdx.x >> 5;
  int lane = threadIdx.x & 31;
  int tile = blockIdx.x * 4 + wave;      // strip index
  constexpr int NTG = G_ / 64;           // 16 strips of 64 cols per m-tile
  int mt = tile / NTG, ng = tile % NTG;
  int m0 = mt * 16, n0 = ng * 64;
  const _Float16* A  = x + (size_t)dir * M_ * KX_;
  const _Float16* Bt = wxT + (size_t)dir * G_ * KX_;
  v8f acc0{}, acc1{}, acc2{}, acc3{};
  int ks = kAct >> 5;
  for (int s = 0; s < ks; ++s) {
    int k0 = s * 32;
    v16h a  = ldA(A, KX_, m0, k0, lane);
    v16h b0 = ldB(Bt, KX_, n0 +  0, k0, lane);
    v16h b1 = ldB(Bt, KX_, n0 + 16, k0, lane);
    v16h b2 = ldB(Bt, KX_, n0 + 32, k0, lane);
    v16h b3 = ldB(Bt, KX_, n0 + 48, k0, lane);
    acc0 = __builtin_amdgcn_wmma_f32_16x16x32_f16(false, a, false, b0, (short)0, acc0, false, false);
    acc1 = __builtin_amdgcn_wmma_f32_16x16x32_f16(false, a, false, b1, (short)0, acc1, false, false);
    acc2 = __builtin_amdgcn_wmma_f32_16x16x32_f16(false, a, false, b2, (short)0, acc2, false, false);
    acc3 = __builtin_amdgcn_wmma_f32_16x16x32_f16(false, a, false, b3, (short)0, acc3, false, false);
  }
  const float* bias = dir ? biasB : biasF;
  _Float16* Cd = xproj + (size_t)dir * M_ * G_;
  int col = lane & 15, rbase = (lane >> 4) * 8;
  v8f accs[4] = {acc0, acc1, acc2, acc3};
#pragma unroll
  for (int q = 0; q < 4; ++q) {
    int n = n0 + q * 16 + col;
    float bv = bias[n];
#pragma unroll
    for (int r = 0; r < 8; ++r) {
      int row = m0 + rbase + r;
      Cd[(size_t)row * G_ + n] = (_Float16)(accs[q][r] + bv);
    }
  }
}

__global__ void k_zero_state(float* __restrict__ cst, float* __restrict__ hst,
                             _Float16* __restrict__ h16) {
  int i = blockIdx.x * blockDim.x + threadIdx.x;
  if (i < 2 * B_ * H_) { cst[i] = 0.0f; hst[i] = 0.0f; }
  if (i < 4 * B_ * H_) h16[i] = (_Float16)0.0f;
}

// One LSTM timestep, both directions (blockIdx.z). Per block: 16 batch rows x
// 32 h-columns; 4 waves = the 4 gates, each wave computes a 16x32 strip of its
// gate (A-tile reused across 2 WMMAs). Gate fusion in LDS.
__global__ __launch_bounds__(128) void k_step(int t, const int* __restrict__ length,
                                              const _Float16* __restrict__ xproj,
                                              const _Float16* __restrict__ whT,
                                              float* __restrict__ cst, float* __restrict__ hst,
                                              _Float16* __restrict__ h16,
                                              float* __restrict__ hseq) {
  int dir  = blockIdx.z;
  int g    = threadIdx.x >> 5;      // gate: 0=i 1=j 2=f 3=o
  int lane = threadIdx.x & 31;
  int m0   = blockIdx.x * 16;       // batch tile
  int nk0  = blockIdx.y * 32;       // h-column strip
  const _Float16* A    = h16 + ((size_t)dir * 2 + (t & 1)) * (B_ * H_);
  _Float16*       Aout = h16 + ((size_t)dir * 2 + ((t + 1) & 1)) * (B_ * H_);
  const _Float16* Bt   = whT + (size_t)dir * G_ * H_;
  v8f acc0{}, acc1{};
#pragma unroll
  for (int s = 0; s < H_ / 32; ++s) {
    int k0 = s * 32;
    v16h a  = ldA(A,  H_, m0, k0, lane);
    v16h b0 = ldB(Bt, H_, g * H_ + nk0 +  0, k0, lane);
    v16h b1 = ldB(Bt, H_, g * H_ + nk0 + 16, k0, lane);
    acc0 = __builtin_amdgcn_wmma_f32_16x16x32_f16(false, a, false, b0, (short)0, acc0, false, false);
    acc1 = __builtin_amdgcn_wmma_f32_16x16x32_f16(false, a, false, b1, (short)0, acc1, false, false);
  }
  __shared__ float zs[4][16][32];
  const _Float16* XP = xproj + (size_t)dir * M_ * G_;
  int col = lane & 15, rbase = (lane >> 4) * 8;
  v8f accs[2] = {acc0, acc1};
#pragma unroll
  for (int q = 0; q < 2; ++q) {
#pragma unroll
    for (int r = 0; r < 8; ++r) {
      int row = rbase + r;
      int gb  = m0 + row;
      float xpv = (float)XP[((size_t)gb * T_ + t) * G_ + g * H_ + nk0 + q * 16 + col];
      zs[g][row][q * 16 + col] = accs[q][r] + xpv;
    }
  }
  __syncthreads();
  for (int e = threadIdx.x; e < 16 * 32; e += 128) {
    int row = e >> 5, kc = e & 31;
    int gb = m0 + row, gk = nk0 + kc;
    float zi = zs[0][row][kc], zj = zs[1][row][kc];
    float zf = zs[2][row][kc], zo = zs[3][row][kc];
    size_t si = (size_t)dir * B_ * H_ + (size_t)gb * H_ + gk;
    float cold = cst[si], hold = hst[si];
    float cnew = sigm(zf + 1.0f) * cold + sigm(zi) * tanhf(zj);
    float hnew = sigm(zo) * tanhf(cnew);
    bool  mv   = t < length[gb];
    float c2 = mv ? cnew : cold;
    float h2 = mv ? hnew : hold;
    cst[si] = c2;
    hst[si] = h2;
    Aout[(size_t)gb * H_ + gk] = (_Float16)h2;
    hseq[(size_t)dir * M_ * H_ + ((size_t)gb * T_ + t) * H_ + gk] = mv ? hnew : 0.0f;
  }
}

// Build next layer input: concat(fw, un-reversed bw) -> f16; optionally also
// the time-reversed copy for the backward pass.
__global__ void k_concat(const int* __restrict__ length, const float* __restrict__ hseq,
                         _Float16* __restrict__ xF, _Float16* __restrict__ xR, int writeRev) {
  int idx = blockIdx.x * blockDim.x + threadIdx.x;
  if (idx >= B_ * T_ * 2 * H_) return;
  int k  = idx % (2 * H_);
  int bt = idx / (2 * H_);
  int t  = bt % T_;
  int b  = bt / T_;
  int L  = length[b];
  int tr = (t < L) ? (L - 1 - t) : t;
  float v;
  if (k < H_) v = hseq[((size_t)b * T_ + t) * H_ + k];
  else        v = hseq[(size_t)M_ * H_ + ((size_t)b * T_ + tr) * H_ + (k - H_)];
  _Float16 hv = (_Float16)v;
  xF[((size_t)b * T_ + t) * KX_ + k] = hv;
  if (writeRev) xR[((size_t)b * T_ + tr) * KX_ + k] = hv;
}

// res_W (2H, D) fp32 -> f16 transposed [NR][KX] (rows >= D zero)
__global__ void k_resw(const float* __restrict__ resW, _Float16* __restrict__ rwT) {
  int idx = blockIdx.x * blockDim.x + threadIdx.x;
  if (idx >= NR_ * KX_) return;
  int k = idx % KX_, n = idx / KX_;
  float v = (n < D_) ? resW[(size_t)k * D_ + n] : 0.0f;
  rwT[(size_t)n * KX_ + k] = (_Float16)v;
}

// lstm_out[m][n] = emb32[m][n] + x2[m][:] @ rwT[n][:] + res_b[n]
// 16x64 strip per wave (WMMA).
__global__ __launch_bounds__(128) void k_resgemm(const _Float16* __restrict__ x2,
                                                 const _Float16* __restrict__ rwT,
                                                 const float* __restrict__ resb,
                                                 const float* __restrict__ emb32,
                                                 float* __restrict__ lout) {
  int wave = threadIdx.x >> 5;
  int lane = threadIdx.x & 31;
  int tile = blockIdx.x * 4 + wave;
  constexpr int NTG = NR_ / 64;     // 6
  int mt = tile / NTG, ng = tile % NTG;
  int m0 = mt * 16, n0 = ng * 64;
  v8f acc0{}, acc1{}, acc2{}, acc3{};
#pragma unroll
  for (int s = 0; s < KX_ / 32; ++s) {
    int k0 = s * 32;
    v16h a  = ldA(x2, KX_, m0, k0, lane);
    v16h b0 = ldB(rwT, KX_, n0 +  0, k0, lane);
    v16h b1 = ldB(rwT, KX_, n0 + 16, k0, lane);
    v16h b2 = ldB(rwT, KX_, n0 + 32, k0, lane);
    v16h b3 = ldB(rwT, KX_, n0 + 48, k0, lane);
    acc0 = __builtin_amdgcn_wmma_f32_16x16x32_f16(false, a, false, b0, (short)0, acc0, false, false);
    acc1 = __builtin_amdgcn_wmma_f32_16x16x32_f16(false, a, false, b1, (short)0, acc1, false, false);
    acc2 = __builtin_amdgcn_wmma_f32_16x16x32_f16(false, a, false, b2, (short)0, acc2, false, false);
    acc3 = __builtin_amdgcn_wmma_f32_16x16x32_f16(false, a, false, b3, (short)0, acc3, false, false);
  }
  int col = lane & 15, rbase = (lane >> 4) * 8;
  v8f accs[4] = {acc0, acc1, acc2, acc3};
#pragma unroll
  for (int q = 0; q < 4; ++q) {
    int n = n0 + q * 16 + col;
    if (n < D_) {
      float bv = resb[n];
#pragma unroll
      for (int r = 0; r < 8; ++r) {
        int row = m0 + rbase + r;
        lout[(size_t)row * D_ + n] = accs[q][r] + bv + emb32[(size_t)row * D_ + n];
      }
    }
  }
}

__global__ void k_crfin(const float* __restrict__ lout, const float* __restrict__ crfW,
                        const float* __restrict__ crfb, float* __restrict__ crfin) {
  int idx = blockIdx.x * blockDim.x + threadIdx.x;
  if (idx >= M_ * 2) return;
  int c = idx & 1, m = idx >> 1;
  float s = crfb[c];
  for (int d = 0; d < D_; ++d) s += lout[(size_t)m * D_ + d] * crfW[d * 2 + c];
  crfin[m * 2 + c] = s;
}

// 2-state CRF forward scan; one thread per batch element.
__global__ void k_crf(const int* __restrict__ length, const float* __restrict__ crfin,
                      const float* __restrict__ trans, float* __restrict__ crfout) {
  int b = blockIdx.x * blockDim.x + threadIdx.x;
  if (b >= B_) return;
  int L = length[b];
  float t00 = trans[0], t01 = trans[1], t10 = trans[2], t11 = trans[3];
  float a0 = crfin[((size_t)b * T_) * 2 + 0];
  float a1 = crfin[((size_t)b * T_) * 2 + 1];
  {
    bool m = 0 < L;
    float mx = fmaxf(a0, a1);
    float e0 = expf(a0 - mx), e1 = expf(a1 - mx);
    crfout[(size_t)b * T_] = m ? (e1 / (e0 + e1)) : 0.0f;
  }
  for (int t = 1; t < T_; ++t) {
    float em0 = crfin[((size_t)b * T_ + t) * 2 + 0];
    float em1 = crfin[((size_t)b * T_ + t) * 2 + 1];
    float x0 = a0 + t00, y0 = a1 + t10;
    float mx0 = fmaxf(x0, y0);
    float n0v = mx0 + logf(expf(x0 - mx0) + expf(y0 - mx0)) + em0;
    float x1 = a0 + t01, y1 = a1 + t11;
    float mx1 = fmaxf(x1, y1);
    float n1v = mx1 + logf(expf(x1 - mx1) + expf(y1 - mx1)) + em1;
    bool m = t < L;
    if (m) { a0 = n0v; a1 = n1v; }
    float mx = fmaxf(a0, a1);
    float e0 = expf(a0 - mx), e1 = expf(a1 - mx);
    crfout[(size_t)b * T_ + t] = m ? (e1 / (e0 + e1)) : 0.0f;
  }
}

__global__ void k_sent(const float* __restrict__ crfout, const float* __restrict__ lout,
                       float* __restrict__ sent) {
  int idx = blockIdx.x * blockDim.x + threadIdx.x;
  if (idx >= B_ * D_) return;
  int d = idx % D_, b = idx / D_;
  float s = 0.0f;
  for (int t = 0; t < T_; ++t)
    s += crfout[(size_t)b * T_ + t] * lout[((size_t)b * T_ + t) * D_ + d];
  sent[(size_t)b * D_ + d] = s;
}

__global__ void k_logits(const float* __restrict__ sent, const float* __restrict__ outW,
                         const float* __restrict__ outb, float* __restrict__ out) {
  int b = blockIdx.x * blockDim.x + threadIdx.x;
  if (b >= B_) return;
  float z[NC_];
  for (int c = 0; c < NC_; ++c) z[c] = outb[c];
  for (int d = 0; d < D_; ++d) {
    float sv = sent[(size_t)b * D_ + d];
    for (int c = 0; c < NC_; ++c) z[c] += sv * outW[d * NC_ + c];
  }
  float mx = fmaxf(z[0], fmaxf(z[1], z[2]));
  float e0 = expf(z[0] - mx), e1 = expf(z[1] - mx), e2 = expf(z[2] - mx);
  float inv = 1.0f / (e0 + e1 + e2);
  out[b * NC_ + 0] = e0 * inv;
  out[b * NC_ + 1] = e1 * inv;
  out[b * NC_ + 2] = e2 * inv;
}

// ---------------- host launcher ----------------
extern "C" void kernel_launch(void* const* d_in, const int* in_sizes, int n_in,
                              void* d_out, int out_size, void* d_ws, size_t ws_size,
                              hipStream_t stream) {
  (void)in_sizes; (void)n_in; (void)out_size;
  if (ws_size < WS_TOTAL) return;

  const int*   tokens    = (const int*)d_in[0];
  const int*   masks     = (const int*)d_in[1];
  const int*   length    = (const int*)d_in[2];
  const float* embedding = (const float*)d_in[3];
  const float* mask_emb  = (const float*)d_in[4];
  const float* transition= (const float*)d_in[5];
  const float* l0fW = (const float*)d_in[6];  const float* l0fb = (const float*)d_in[7];
  const float* l0bW = (const float*)d_in[8];  const float* l0bb = (const float*)d_in[9];
  const float* l1fW = (const float*)d_in[10]; const float* l1fb = (const float*)d_in[11];
  const float* l1bW = (const float*)d_in[12]; const float* l1bb = (const float*)d_in[13];
  const float* resW = (const float*)d_in[14]; const float* resb = (const float*)d_in[15];
  const float* crfW = (const float*)d_in[16]; const float* crfb = (const float*)d_in[17];
  const float* outW = (const float*)d_in[18]; const float* outb = (const float*)d_in[19];
  float* out = (float*)d_out;

  char* w = (char*)d_ws;
  float*    emb32  = (float*)(w + O_EMB);
  _Float16* xF     = (_Float16*)(w + O_X);
  _Float16* xR     = (_Float16*)(w + O_X + SZ_X);
  _Float16* xproj  = (_Float16*)(w + O_XP);
  float*    hseq   = (float*)(w + O_HS);
  _Float16* wxT    = (_Float16*)(w + O_WXT);
  _Float16* whT    = (_Float16*)(w + O_WHT);
  _Float16* rwT    = (_Float16*)(w + O_RWT);
  float*    lout   = (float*)(w + O_LO);
  float*    crfin  = (float*)(w + O_CIN);
  float*    crfout = (float*)(w + O_COUT);
  float*    sent   = (float*)(w + O_SENT);
  float*    cst    = (float*)(w + O_CST);
  float*    hst    = (float*)(w + O_HST);
  _Float16* h16    = (_Float16*)(w + O_H16);

  // 1) embedding gather (also builds f16 forward + reversed layer0 inputs)
  hipLaunchKernelGGL(k_embed, dim3((B_ * T_ * K0_) / 256), dim3(256), 0, stream,
                     tokens, masks, length, embedding, mask_emb, emb32, xF, xR);

  // 2) two BiLSTM layers
  for (int layer = 0; layer < 2; ++layer) {
    const float* Wf = layer ? l1fW : l0fW;
    const float* Wb = layer ? l1bW : l0bW;
    const float* bf = layer ? l1fb : l0fb;
    const float* bb = layer ? l1bb : l0bb;
    int Din  = layer ? 2 * H_ : D_;
    int kAct = layer ? KX_ : K0_;

    hipLaunchKernelGGL(k_wprep, dim3((2 * G_ * KX_) / 256), dim3(256), 0, stream,
                       Wf, Wb, Din, wxT, whT);
    hipLaunchKernelGGL(k_xproj, dim3((M_ / 16) * (G_ / 64) / 4, 1, 2), dim3(128), 0, stream,
                       xF, wxT, bf, bb, xproj, kAct);
    hipLaunchKernelGGL(k_zero_state, dim3((4 * B_ * H_) / 256), dim3(256), 0, stream,
                       cst, hst, h16);
    for (int t = 0; t < T_; ++t)
      hipLaunchKernelGGL(k_step, dim3(B_ / 16, H_ / 32, 2), dim3(128), 0, stream,
                         t, length, xproj, whT, cst, hst, h16, hseq);
    // layer0 -> build layer1 inputs (fwd + reversed); layer1 -> build x2 (fwd only)
    hipLaunchKernelGGL(k_concat, dim3((B_ * T_ * 2 * H_) / 256), dim3(256), 0, stream,
                       length, hseq, xF, xR, layer == 0 ? 1 : 0);
  }

  // 3) residual projection (WMMA) + head
  hipLaunchKernelGGL(k_resw, dim3((NR_ * KX_) / 256), dim3(256), 0, stream, resW, rwT);
  hipLaunchKernelGGL(k_resgemm, dim3((M_ / 16) * (NR_ / 64) / 4), dim3(128), 0, stream,
                     xF, rwT, resb, emb32, lout);
  hipLaunchKernelGGL(k_crfin, dim3((M_ * 2) / 256), dim3(256), 0, stream,
                     lout, crfW, crfb, crfin);
  hipLaunchKernelGGL(k_crf, dim3(1), dim3(64), 0, stream, length, crfin, transition, crfout);
  hipLaunchKernelGGL(k_sent, dim3((B_ * D_ + 255) / 256), dim3(256), 0, stream,
                     crfout, lout, sent);
  hipLaunchKernelGGL(k_logits, dim3(1), dim3(64), 0, stream, sent, outW, outb, out);
}
